// GPRModel_19387482374798
// MI455X (gfx1250) — matrix operand order repaired and checked by reading, not compiled, not run on hardware
//
#include <hip/hip_runtime.h>
#include <hip/hip_bf16.h>
#include <math.h>

// ---- problem constants (match reference) ----
#define BB   64
#define NN   256
#define NTOT 16384      // BB*NN
#define EDG  524288
#define EPG  8192       // edges per graph = NN * DEG(32)
#define HD   1024
#define OC   256
#define RR   3
#define BERT_IN  3072
#define BERT_OUT 1536
#define D1   2304
#define EPSV 1e-5f
#define NEGV (-1e12f)

typedef __bf16 bf16_t;
typedef bf16_t v8bf  __attribute__((ext_vector_type(8)));
typedef bf16_t v16bf __attribute__((ext_vector_type(16)));
typedef float  v8f   __attribute__((ext_vector_type(8)));

__device__ __forceinline__ unsigned short f32_to_bf16_rne(float f) {
  unsigned int u = __float_as_uint(f);
  unsigned int r = 0x7FFFu + ((u >> 16) & 1u);
  return (unsigned short)((u + r) >> 16);
}

// A fragment (16x32 bf16): two contiguous 16B loads per lane
__device__ __forceinline__ v16bf ldA(const unsigned short* p) {
  v8bf lo = *(const v8bf*)p;
  v8bf hi = *(const v8bf*)(p + 8);
  return __builtin_shufflevector(lo, hi,
      0, 1, 2, 3, 4, 5, 6, 7, 8, 9, 10, 11, 12, 13, 14, 15);
}

__device__ __forceinline__ v8f wmma_bf16(v16bf a, v16bf b, v8f c) {
  return __builtin_amdgcn_wmma_f32_16x16x32_bf16(
      false, a, false, b, (short)0, c, false, false);
}

__device__ __forceinline__ void stD(float* base, v8f acc) {
#pragma unroll
  for (int j = 0; j < 8; ++j) base[(size_t)j * OC] = acc[j];
}

// ---------------- conversion kernels ----------------
__global__ void k_cvt_h(const float* __restrict__ h, unsigned short* __restrict__ o, int n) {
  int i = blockIdx.x * 256 + threadIdx.x;
  if (i < n) o[i] = f32_to_bf16_rne(h[i]);
}

// W[r][k][n] (f32) -> Wt[r][n][k] (bf16) : column-major B for WMMA
__global__ void k_cvt_wt(const float* __restrict__ W, unsigned short* __restrict__ Wt, int n) {
  int i = blockIdx.x * 256 + threadIdx.x;
  if (i >= n) return;
  int k  = i & (HD - 1);
  int t  = i >> 10;
  int nn = t & (OC - 1);
  int r  = t >> 8;
  Wt[i] = f32_to_bf16_rne(W[((size_t)r * HD + k) * OC + nn]);
}

// ---------------- RGCN GEMM: Hm[r] = h(16384x1024) @ W[r](1024x256) ----------------
// bf16 WMMA, f32 acc. One wave computes a 32x64 output block (2 M-tiles x 4 N-tiles,
// 8 accumulators) for ~2.7x less fragment traffic than 1 tile/wave.
// Grid exactly 6144 waves -> EXEC all-1s everywhere (WMMA requirement).
__global__ void __launch_bounds__(256)
k_rgcn_gemm(const unsigned short* __restrict__ hbf,   // [NTOT][HD] bf16
            const unsigned short* __restrict__ Wt,    // [RR][OC][HD] bf16 (col-major B)
            float* __restrict__ Hm)                   // [RR][NTOT][OC] f32
{
  const int gtid = blockIdx.x * 256 + threadIdx.x;
  const int wave = gtid >> 5;
  const int lane = gtid & 31;
  const int mb = wave & 511;          // 512 M-blocks of 32 rows
  const int t  = wave >> 9;
  const int nb = t & 3;               // 4 N-blocks of 64 cols
  const int r  = t >> 2;              // 0..2
  const int hi  = lane >> 4;
  const int l15 = lane & 15;
  const int m0 = mb * 32;
  const int n0 = nb * 64;

  const unsigned short* pa0 = hbf + (size_t)(m0 + l15) * HD + hi * 16;
  const unsigned short* pa1 = pa0 + (size_t)16 * HD;
  const unsigned short* pb0 = Wt + ((size_t)r * OC + n0 + l15) * HD + hi * 16;
  const unsigned short* pb1 = pb0 + (size_t)16 * HD;
  const unsigned short* pb2 = pb0 + (size_t)32 * HD;
  const unsigned short* pb3 = pb0 + (size_t)48 * HD;

  v8f a00 = {}, a01 = {}, a02 = {}, a03 = {};
  v8f a10 = {}, a11 = {}, a12 = {}, a13 = {};

#pragma unroll 2
  for (int k0 = 0; k0 < HD; k0 += 32) {
    __builtin_prefetch(pa0 + k0 + 128, 0, 0);
    __builtin_prefetch(pa1 + k0 + 128, 0, 0);
    v16bf A0 = ldA(pa0 + k0);
    v16bf A1 = ldA(pa1 + k0);
    v16bf B0 = *(const v16bf*)(pb0 + k0);
    v16bf B1 = *(const v16bf*)(pb1 + k0);
    v16bf B2 = *(const v16bf*)(pb2 + k0);
    v16bf B3 = *(const v16bf*)(pb3 + k0);
    a00 = wmma_bf16(A0, B0, a00);
    a01 = wmma_bf16(A0, B1, a01);
    a02 = wmma_bf16(A0, B2, a02);
    a03 = wmma_bf16(A0, B3, a03);
    a10 = wmma_bf16(A1, B0, a10);
    a11 = wmma_bf16(A1, B1, a11);
    a12 = wmma_bf16(A1, B2, a12);
    a13 = wmma_bf16(A1, B3, a13);
  }

  // D layout: VGPR j -> row (j + 8*hi), col l15 within each 16x16 tile
  float* d0 = Hm + ((size_t)r * NTOT + m0 + hi * 8) * OC + n0 + l15;
  stD(d0,      a00); stD(d0 + 16, a01); stD(d0 + 32, a02); stD(d0 + 48, a03);
  float* d1 = d0 + (size_t)16 * OC;
  stD(d1,      a10); stD(d1 + 16, a11); stD(d1 + 32, a12); stD(d1 + 48, a13);
}

// ---------------- gate GEMV: Gm[r][n] = dot(h[n], GW[r]) (f32) ----------------
__global__ void __launch_bounds__(256)
k_gate(const float* __restrict__ h, const float* __restrict__ GW, float* __restrict__ Gm) {
  int gtid = blockIdx.x * 256 + threadIdx.x;
  int wave = gtid >> 5, lane = gtid & 31;
  int n = wave & (NTOT - 1);
  int r = wave >> 14;                   // exactly 0..2
  const float* hp = h + (size_t)n * HD;
  const float* gp = GW + (size_t)r * HD;
  float acc = 0.f;
  for (int k = lane; k < HD; k += 32) acc = fmaf(hp[k], gp[k], acc);
  for (int o = 16; o > 0; o >>= 1) acc += __shfl_down(acc, o, 32);
  if (lane == 0) Gm[(size_t)r * NTOT + n] = acc;
}

// ---------------- per-graph fused kernel: edge scatter (LDS atomics) + relu + pooling ----
// One workgroup per graph. The graph accumulator (256 nodes x 256 ch x 4B = 256KB)
// fits in the 320KB CDNA5 WGP LDS -> ds_add_f32 instead of global atomics, and
// pooling reads straight from LDS (no hn global round trip, no zero kernel).
__global__ void __launch_bounds__(1024)
k_graph(const int* __restrict__ src, const int* __restrict__ dst, const int* __restrict__ rel,
        const float* __restrict__ norm, const float* __restrict__ Gm,
        const float* __restrict__ Hm,
        const unsigned char* __restrict__ e1, const unsigned char* __restrict__ e2,
        float* __restrict__ emb) {
  extern __shared__ float lhn[];        // [NN][OC] = 256KB
  const int b   = blockIdx.x;
  const int tid = threadIdx.x;

  // zero the LDS accumulator (fresh every launch -> deterministic)
  for (int i = tid; i < NN * OC; i += 1024) lhn[i] = 0.f;
  __syncthreads();

  // scatter: 32 waves x 256 edges each
  const int lane = tid & 31;
  const int wv   = tid >> 5;
  const int ebase = b * EPG + wv * (EPG / 32);
  for (int t = 0; t < EPG / 32; ++t) {
    int e = ebase + t;
    int s = src[e], d = dst[e], r = rel[e];
    float g = Gm[(size_t)r * NTOT + s];
    float scale = norm[e] / (1.f + expf(-g));
    const float* m = Hm + ((size_t)r * NTOT + s) * OC;
    float* o = lhn + (size_t)(d - b * NN) * OC;   // dst stays within its own graph
#pragma unroll
    for (int j = 0; j < OC / 32; ++j)
      atomicAdd(o + lane + j * 32, m[lane + j * 32] * scale);  // ds_add_f32
  }
  __syncthreads();

  // relu + masked max / mean pooling -> emb[:, 0:768]
  if (tid < OC) {
    const int c = tid;
    float mx = -INFINITY, s1 = 0.f, s2 = 0.f;
    int c1 = 0, c2 = 0;
    for (int n = 0; n < NN; ++n) {
      float x = lhn[n * OC + c];
      x = x > 0.f ? x : 0.f;                       // relu
      unsigned char m1 = e1[b * NN + n], m2 = e2[b * NN + n];
      float tv = ((!m1) || (!m2)) ? NEGV : x;      // token_mask = (~e1)|(~e2) -> NEG
      mx = fmaxf(mx, tv);
      if (!m1) { s1 += x; ++c1; }
      if (!m2) { s2 += x; ++c2; }
    }
    float* eb = emb + (size_t)b * D1;
    eb[c]          = mx;
    eb[OC + c]     = s1 / (float)c1;
    eb[2 * OC + c] = s2 / (float)c2;
  }
}

// ---------------- train-mode BatchNorm1d over BB rows (biased var) ----------------
__global__ void k_bn(const float* __restrict__ x, const float* __restrict__ gamma,
                     const float* __restrict__ beta, float* __restrict__ y, int cols) {
  int j = blockIdx.x * blockDim.x + threadIdx.x;
  if (j >= cols) return;
  float s = 0.f, ss = 0.f;
  for (int b = 0; b < BB; ++b) { float v = x[(size_t)b * cols + j]; s += v; ss += v * v; }
  float mu  = s * (1.f / BB);
  float var = ss * (1.f / BB) - mu * mu;
  float inv = rsqrtf(var + EPSV);
  float ga = gamma[j], be = beta[j];
  for (int b = 0; b < BB; ++b)
    y[(size_t)b * cols + j] = ga * (x[(size_t)b * cols + j] - mu) * inv + be;
}

// ---------------- small dense layer, 8 batch rows per thread (8x less W traffic) ----------
__global__ void k_linear8(const float* __restrict__ x, const float* __restrict__ W,
                          const float* __restrict__ bias, float* __restrict__ y,
                          int K, int Nout, int ldy, int doRelu) {
  int idx = blockIdx.x * blockDim.x + threadIdx.x;
  if (idx >= (BB / 8) * Nout) return;
  int o  = idx % Nout;
  int bg = idx / Nout;                  // batch group of 8 rows
  const float* xr = x + (size_t)bg * 8 * K;
  float bias_o = bias[o];
  float acc[8];
#pragma unroll
  for (int i = 0; i < 8; ++i) acc[i] = bias_o;
  for (int k = 0; k < K; ++k) {
    float w = W[(size_t)k * Nout + o];
#pragma unroll
    for (int i = 0; i < 8; ++i) acc[i] = fmaf(xr[(size_t)i * K + k], w, acc[i]);
  }
#pragma unroll
  for (int i = 0; i < 8; ++i) {
    float v = doRelu ? fmaxf(acc[i], 0.f) : acc[i];
    y[(size_t)(bg * 8 + i) * ldy + o] = v;
  }
}

// ---------------- host-side launch ----------------
extern "C" void kernel_launch(void* const* d_in, const int* in_sizes, int n_in,
                              void* d_out, int out_size, void* d_ws, size_t ws_size,
                              hipStream_t stream) {
  const float* h    = (const float*)d_in[0];
  const int*   src  = (const int*)d_in[1];
  const int*   dst  = (const int*)d_in[2];
  const int*   rel  = (const int*)d_in[3];
  const float* norm = (const float*)d_in[4];
  const unsigned char* e1m = (const unsigned char*)d_in[5];
  const unsigned char* e2m = (const unsigned char*)d_in[6];
  const float* bert = (const float*)d_in[7];
  const float* W    = (const float*)d_in[8];
  const float* GW   = (const float*)d_in[9];
  const float* bh_gamma = (const float*)d_in[10];
  const float* bh_beta  = (const float*)d_in[11];
  const float* bh_W     = (const float*)d_in[12];
  const float* bh_b     = (const float*)d_in[13];
  const float* hd_g1  = (const float*)d_in[14];
  const float* hd_be1 = (const float*)d_in[15];
  const float* hd_W1  = (const float*)d_in[16];
  const float* hd_b1  = (const float*)d_in[17];
  const float* hd_g2  = (const float*)d_in[18];
  const float* hd_be2 = (const float*)d_in[19];
  const float* hd_W2  = (const float*)d_in[20];
  const float* hd_b2  = (const float*)d_in[21];
  float* out = (float*)d_out;

  // ---- workspace carve (all chunks multiple of 256B) ----
  char* w = (char*)d_ws;
  unsigned short* hbf = (unsigned short*)w; w += (size_t)NTOT * HD * 2;        // 32 MB
  unsigned short* Wt  = (unsigned short*)w; w += (size_t)RR * OC * HD * 2;     // 1.5 MB
  float* Hm   = (float*)w; w += (size_t)RR * NTOT * OC * 4;                    // 48 MB
  float* Gm   = (float*)w; w += (size_t)RR * NTOT * 4;                         // 192 KB
  float* xbn  = (float*)w; w += (size_t)BB * BERT_IN * 4;                      // 768 KB
  float* emb  = (float*)w; w += (size_t)BB * D1 * 4;                           // 576 KB
  float* embn = (float*)w; w += (size_t)BB * D1 * 4;                           // 576 KB
  float* y1   = (float*)w; w += (size_t)BB * OC * 4;                           // 64 KB
  float* y1n  = (float*)w; w += (size_t)BB * OC * 4;                           // 64 KB

  // 1) bf16 conversions
  k_cvt_h<<<(NTOT * HD) / 256, 256, 0, stream>>>(h, hbf, NTOT * HD);
  k_cvt_wt<<<(RR * OC * HD + 255) / 256, 256, 0, stream>>>(W, Wt, RR * OC * HD);

  // 2) WMMA GEMM Hm = h @ W[r]  (512*4*3 = 6144 waves exactly -> 768 blocks)
  k_rgcn_gemm<<<768, 256, 0, stream>>>(hbf, Wt, Hm);

  // 3) gate GEMV Gm
  k_gate<<<RR * NTOT / 8, 256, 0, stream>>>(h, GW, Gm);

  // 4) fused per-graph edge scatter (LDS atomics) + relu + pooling -> emb[:, :768]
  (void)hipFuncSetAttribute(reinterpret_cast<const void*>(k_graph),
                            hipFuncAttributeMaxDynamicSharedMemorySize, NN * OC * 4);
  k_graph<<<BB, 1024, NN * OC * 4, stream>>>(src, dst, rel, norm, Gm, Hm, e1m, e2m, emb);

  // 5) BERT head: BN -> linear+relu -> emb[:, 768:]
  k_bn<<<(BERT_IN + 255) / 256, 256, 0, stream>>>(bert, bh_gamma, bh_beta, xbn, BERT_IN);
  k_linear8<<<((BB / 8) * BERT_OUT + 255) / 256, 256, 0, stream>>>(
      xbn, bh_W, bh_b, emb + 768, BERT_IN, BERT_OUT, D1, 1);

  // 6) final MLP head
  k_bn<<<(D1 + 255) / 256, 256, 0, stream>>>(emb, hd_g1, hd_be1, embn, D1);
  k_linear8<<<((BB / 8) * OC + 255) / 256, 256, 0, stream>>>(
      embn, hd_W1, hd_b1, y1, D1, OC, OC, 1);
  k_bn<<<1, 256, 0, stream>>>(y1, hd_g2, hd_be2, y1n, OC);
  k_linear8<<<((BB / 8) * 6 + 255) / 256, 256, 0, stream>>>(
      y1n, hd_W2, hd_b2, out, OC, 6, 6, 0);
}
